// Net_74887049773819
// MI455X (gfx1250) — matrix operand order repaired and checked by reading, not compiled, not run on hardware
//
#include <hip/hip_runtime.h>
#include <hip/hip_bf16.h>

typedef __attribute__((ext_vector_type(16))) __bf16 v16bf;
typedef __attribute__((ext_vector_type(8)))  float  v8f;

#define N_NODES 500000
#define N_TILES 31250        // 500000 / 16, exact
#define F_IN    165
#define HID     32

// Packed-weight layout in d_ws (bf16 units):
//   [0, 24576)      : 48 x-weight fragments  ((g*6+t)*2+n)*512
//   [24576, 28672)  : 8  h-weight fragments  (g*2+n)*512
//   [28672, 29184)  : 1  classifier fragment
// Each fragment: 32 lanes * 16 bf16 (lane-major) = 512 entries = 1024 B.
#define WS_BF16_COUNT 29184
#define WS_BYTES      (WS_BF16_COUNT * 2)   // 58368
#define LDS_HW_OFF    49152                 // bytes
#define LDS_LIN_OFF   57344                 // bytes
#define LDS_SCRATCH   WS_BYTES              // per-wave 2KB scratch after weights

// Hardware tanh (CDNA5 TRANS op) with safe fallback.
#if defined(__has_builtin)
#if __has_builtin(__builtin_amdgcn_tanhf)
#define FAST_TANH(v) __builtin_amdgcn_tanhf(v)
#endif
#endif
#ifndef FAST_TANH
#define FAST_TANH(v) tanhf(v)
#endif

__device__ __forceinline__ float sigmoid_t(float v) {
    // sigmoid(x) = 0.5*tanh(x/2) + 0.5
    return __builtin_fmaf(0.5f, FAST_TANH(0.5f * v), 0.5f);
}

// ---------------------------------------------------------------------------
// Kernel 1: pack f32 weights into bf16 WMMA B-fragment layout.
// B-fragment (16-bit, 32x16 K x N): lane holds 16 bf16; entry j corresponds to
//   K = ktile*32 + (lane>=16 ? 8 : 0) + (j & 7) + (j >= 8 ? 16 : 0),  col = ntile*16 + (lane & 15)
// ---------------------------------------------------------------------------
__global__ void pack_weights_kernel(
    const float* Wi, const float* Wf, const float* Wc, const float* Wo,
    const float* Wci, const float* Wcf, const float* Wcc, const float* Wco,
    const float* Wlin, __bf16* ws)
{
    const int b = blockIdx.x;
    for (int e = threadIdx.x; e < 512; e += blockDim.x) {
        const int lane = e >> 4;
        const int j    = e & 15;
        const int k32  = ((lane >> 4) << 3) + (j & 7) + ((j >> 3) << 4);   // K within 32-wide tile
        const int colL = lane & 15;
        float v = 0.0f;
        int widx;
        if (b < 48) {                               // x-weights [165 x 32], K padded to 192
            const int g = b / 12, rem = b % 12, t = rem >> 1, n = rem & 1;
            const float* W = (g == 0) ? Wi : (g == 1) ? Wf : (g == 2) ? Wc : Wo;
            const int K = t * 32 + k32, col = n * 16 + colL;
            if (K < F_IN) v = W[K * HID + col];
            widx = b * 512 + e;
        } else if (b < 56) {                        // h-weights [32 x 32]
            const int bb = b - 48, g = bb >> 1, n = bb & 1;
            const float* W = (g == 0) ? Wci : (g == 1) ? Wcf : (g == 2) ? Wcc : Wco;
            v = W[k32 * HID + (n * 16 + colL)];
            widx = 24576 + bb * 512 + e;
        } else {                                    // classifier [32 x 2], cols 2..15 zero
            if (colL < 2) v = Wlin[k32 * 2 + colL];
            widx = 28672 + e;
        }
        ws[widx] = (__bf16)v;
    }
}

// ---------------------------------------------------------------------------
// Kernel 2: fused GCLSTM cell. 8 wave32 per block, one 16-row tile per wave.
// ---------------------------------------------------------------------------
__global__ void __launch_bounds__(256)
gclstm_kernel(const float* __restrict__ x, const float* __restrict__ h,
              const float* __restrict__ c,
              const float* bi, const float* bf_, const float* bg, const float* bo,
              const float* bci, const float* bcf, const float* bcg, const float* bco,
              const float* blin,
              const __bf16* __restrict__ ws,
              float* __restrict__ y_out, float* __restrict__ h_out,
              float* __restrict__ c_out)
{
    extern __shared__ __align__(16) char smem[];

    // Stage packed weights into LDS via CDNA5 async global->LDS DMA
    // (ASYNCcnt-tracked; no VGPR round trip).
    {
        const unsigned lds_base = (unsigned)(uintptr_t)(void*)smem;
        const unsigned long long gbase = (unsigned long long)(uintptr_t)ws;
        for (int i = threadIdx.x; i < WS_BYTES / 16; i += 256) {
            unsigned           lds_a = lds_base + (unsigned)(i * 16);
            unsigned long long ga    = gbase + (unsigned long long)(i * 16);
            asm volatile("global_load_async_to_lds_b128 %0, %1, off"
                         :: "v"(lds_a), "v"(ga) : "memory");
        }
        asm volatile("s_wait_asynccnt 0" ::: "memory");
    }
    __syncthreads();

    const int wave = threadIdx.x >> 5;
    const int lane = threadIdx.x & 31;
    const int tile = blockIdx.x * 8 + wave;
    if (tile >= N_TILES) return;

    const int half = lane >> 4;        // 0: lanes 0-15, 1: lanes 16-31
    const int colL = lane & 15;
    const int koff = half * 8;         // A-fragment K sub-offset
    const int row0 = tile * 16;

    float* scratch = reinterpret_cast<float*>(smem + LDS_SCRATCH + wave * 2048);

    // ---- A fragment of h (16x32, M = colL; K runs per CDNA5 ISA layout) ----
    v16bf ah;
    {
        const float* hr = h + (size_t)(row0 + colL) * HID;
        const float4* p0 = reinterpret_cast<const float4*>(hr + koff);
        const float4* p1 = reinterpret_cast<const float4*>(hr + 16 + koff);
        float4 q0 = p0[0], q1 = p0[1], q2 = p1[0], q3 = p1[1];
        float av[16] = { q0.x, q0.y, q0.z, q0.w, q1.x, q1.y, q1.z, q1.w,
                         q2.x, q2.y, q2.z, q2.w, q3.x, q3.y, q3.z, q3.w };
#pragma unroll
        for (int j = 0; j < 16; ++j) ah[j] = (__bf16)av[j];
    }

    // ---- accumulators seeded with zero: WMMA SRC2 becomes inline 0 ----
    v8f acc[4][2];
#pragma unroll
    for (int g = 0; g < 4; ++g)
#pragma unroll
        for (int n = 0; n < 2; ++n) acc[g][n] = (v8f){};

    // ---- x @ W_g : 6 K-steps of 32 (165 padded to 192) ----
    const float* xr = x + (size_t)(row0 + colL) * F_IN;
#pragma unroll
    for (int t = 0; t < 6; ++t) {
        v16bf ax;
#pragma unroll
        for (int j = 0; j < 8; ++j) {
            const int k1 = t * 32 + koff + j;
            const int k2 = t * 32 + 16 + koff + j;
            const float v1 = (k1 < F_IN) ? xr[k1] : 0.0f;
            const float v2 = (k2 < F_IN) ? xr[k2] : 0.0f;
            ax[j]     = (__bf16)v1;
            ax[8 + j] = (__bf16)v2;
        }
#pragma unroll
        for (int g = 0; g < 4; ++g)
#pragma unroll
            for (int n = 0; n < 2; ++n) {
                const uint4* wp = reinterpret_cast<const uint4*>(
                    smem + ((g * 6 + t) * 2 + n) * 1024 + lane * 32);
                union { uint4 q[2]; v16bf v; } bu;
                bu.q[0] = wp[0]; bu.q[1] = wp[1];
                acc[g][n] = __builtin_amdgcn_wmma_f32_16x16x32_bf16(
                    false, ax, false, bu.v, (short)0, acc[g][n], false, false);
            }
    }

    // ---- h @ Wc_g : single K-step ----
#pragma unroll
    for (int g = 0; g < 4; ++g)
#pragma unroll
        for (int n = 0; n < 2; ++n) {
            const uint4* wp = reinterpret_cast<const uint4*>(
                smem + LDS_HW_OFF + (g * 2 + n) * 1024 + lane * 32);
            union { uint4 q[2]; v16bf v; } bu;
            bu.q[0] = wp[0]; bu.q[1] = wp[1];
            acc[g][n] = __builtin_amdgcn_wmma_f32_16x16x32_bf16(
                false, ah, false, bu.v, (short)0, acc[g][n], false, false);
        }

    // ---- fused biases (b_g + bc_g), applied in the epilogue ----
    const float* bptr[4]  = { bi, bf_, bg, bo };
    const float* bcptr[4] = { bci, bcf, bcg, bco };
    float bsum[4][2];
#pragma unroll
    for (int g = 0; g < 4; ++g)
#pragma unroll
        for (int n = 0; n < 2; ++n)
            bsum[g][n] = bptr[g][n * 16 + colL] + bcptr[g][n * 16 + colL];

    // ---- elementwise LSTM math on C-layout accumulators ----
    // C/D layout: VGPR r, lanes 0-15 -> (M=r, N=colL); lanes 16-31 -> (M=8+r)
#pragma unroll
    for (int n = 0; n < 2; ++n)
#pragma unroll
        for (int r = 0; r < 8; ++r) {
            const int row = row0 + r + half * 8;
            const int col = n * 16 + colL;
            const size_t idx = (size_t)row * HID + col;
            const float iv = sigmoid_t(acc[0][n][r] + bsum[0][n]);
            const float fv = sigmoid_t(acc[1][n][r] + bsum[1][n]);
            const float gv = FAST_TANH(acc[2][n][r] + bsum[2][n]);
            const float ov = sigmoid_t(acc[3][n][r] + bsum[3][n]);
            const float cv = c[idx];
            const float c0 = fv * cv + iv * gv;
            const float h0 = ov * FAST_TANH(c0);
            c_out[idx] = c0;
            h_out[idx] = h0;
            scratch[(size_t)(r + half * 8) * HID + col] = fmaxf(h0, 0.0f);
        }
    asm volatile("s_wait_dscnt 0" ::: "memory");   // LDS relayout visible in-wave

    // ---- classifier: relu(h0) @ W_lin via one WMMA ----
    v16bf al;
    {
        const float* sr = scratch + (size_t)colL * HID;
        const float4* p0 = reinterpret_cast<const float4*>(sr + koff);
        const float4* p1 = reinterpret_cast<const float4*>(sr + 16 + koff);
        float4 q0 = p0[0], q1 = p0[1], q2 = p1[0], q3 = p1[1];
        float av[16] = { q0.x, q0.y, q0.z, q0.w, q1.x, q1.y, q1.z, q1.w,
                         q2.x, q2.y, q2.z, q2.w, q3.x, q3.y, q3.z, q3.w };
#pragma unroll
        for (int j = 0; j < 16; ++j) al[j] = (__bf16)av[j];
    }
    v8f yacc = (v8f){};
    {
        const uint4* wp = reinterpret_cast<const uint4*>(smem + LDS_LIN_OFF + lane * 32);
        union { uint4 q[2]; v16bf v; } bu;
        bu.q[0] = wp[0]; bu.q[1] = wp[1];
        yacc = __builtin_amdgcn_wmma_f32_16x16x32_bf16(
            false, al, false, bu.v, (short)0, yacc, false, false);
    }
    if (colL < 2) {
        const float bl = blin[colL];
#pragma unroll
        for (int r = 0; r < 8; ++r) {
            const int row = row0 + r + half * 8;
            y_out[(size_t)row * 2 + colL] = yacc[r] + bl;
        }
    }
}

// ---------------------------------------------------------------------------
extern "C" void kernel_launch(void* const* d_in, const int* in_sizes, int n_in,
                              void* d_out, int out_size, void* d_ws, size_t ws_size,
                              hipStream_t stream) {
    const float* x    = (const float*)d_in[0];
    // d_in[1] edge_index, d_in[2] edge_weight: unused (ChebConv K=1)
    const float* h    = (const float*)d_in[3];
    const float* c    = (const float*)d_in[4];
    const float* W_i  = (const float*)d_in[5];
    const float* W_f  = (const float*)d_in[6];
    const float* W_c  = (const float*)d_in[7];
    const float* W_o  = (const float*)d_in[8];
    const float* b_i  = (const float*)d_in[9];
    const float* b_f  = (const float*)d_in[10];
    const float* b_c  = (const float*)d_in[11];
    const float* b_o  = (const float*)d_in[12];
    const float* Wc_i = (const float*)d_in[13];
    const float* Wc_f = (const float*)d_in[14];
    const float* Wc_c = (const float*)d_in[15];
    const float* Wc_o = (const float*)d_in[16];
    const float* bc_i = (const float*)d_in[17];
    const float* bc_f = (const float*)d_in[18];
    const float* bc_c = (const float*)d_in[19];
    const float* bc_o = (const float*)d_in[20];
    const float* Wlin = (const float*)d_in[21];
    const float* blin = (const float*)d_in[22];

    float* out   = (float*)d_out;
    float* y_out = out;                        // [N, 2]
    float* h_out = out + (size_t)N_NODES * 2;  // [N, 32]
    float* c_out = h_out + (size_t)N_NODES * HID;

    __bf16* ws = (__bf16*)d_ws;

    pack_weights_kernel<<<57, 256, 0, stream>>>(
        W_i, W_f, W_c, W_o, Wc_i, Wc_f, Wc_c, Wc_o, Wlin, ws);

    const int blocks = (N_TILES + 7) / 8;      // 3907
    const size_t lds = WS_BYTES + 8 * 2048;    // 74752 B
    gclstm_kernel<<<blocks, 256, lds, stream>>>(
        x, h, c, b_i, b_f, b_c, b_o, bc_i, bc_f, bc_c, bc_o, blin,
        ws, y_out, h_out, c_out);
}
